// TokenHead_MFS_6425271075416
// MI455X (gfx1250) — matrix operand order repaired and checked by reading, not compiled, not run on hardware
//
#include <hip/hip_runtime.h>
#include <stdint.h>

#define NDIM 1024
#define DDIM 1024
#define VDIM 32000
#define KFAC 8

#define MTILE 64
#define VTILE 256
#define KCH   32
#define NCHUNK (DDIM / KCH)        // 32 chunks per facet
#define TCHUNK (KFAC * NCHUNK)     // 256 total
#define XPAD  36                   // f32 per x LDS row (conflict-free, 16B granules)
#define XBUF  (MTILE * XPAD)       // f32 per x buffer
#define WPADB 40                   // bf16 per W LDS row (80B: 16B-aligned, stride 20 banks)
#define WBUFB (VTILE * WPADB)      // bf16 per W buffer

typedef __bf16 bf16_t;
typedef __attribute__((ext_vector_type(16))) __bf16 v16bf;
typedef __attribute__((ext_vector_type(8)))  __bf16 v8bf;
typedef __attribute__((ext_vector_type(8)))  float  v8f;
typedef __attribute__((ext_vector_type(4)))  float  v4f;

// ---- gfx1250 async global->LDS DMA (ASYNCcnt-tracked), 16B per lane ----
__device__ __forceinline__ void async_ld16(uint32_t lds_off, const float* gptr) {
    asm volatile("global_load_async_to_lds_b128 %0, %1, off"
                 :: "v"(lds_off), "v"(gptr) : "memory");
}
__device__ __forceinline__ void wait_async0() {
    asm volatile("s_wait_asynccnt 0x0" ::: "memory");
}

// A fragment: scale by gate and convert (one row per lane)
__device__ __forceinline__ v16bf cvt16s(v8f lo, v8f hi, float s) {
    v16bf r;
#pragma unroll
    for (int i = 0; i < 8; ++i) {
        r[i]     = (bf16_t)(lo[i] * s);
        r[i + 8] = (bf16_t)(hi[i] * s);
    }
    return r;
}

// ---------------- workspace init ----------------
__global__ void init_ws_kernel(float* gsum) {
    if (threadIdx.x == 0) *gsum = 0.0f;
}

// ---------------- gate: g = softmax(x @ Wg^T + bg) ----------------
__global__ __launch_bounds__(256) void gate_kernel(
    const float* __restrict__ x, const float* __restrict__ Wg,
    const float* __restrict__ bg, float* __restrict__ g,
    float* __restrict__ gsum)
{
    const int lane = threadIdx.x & 31;
    const int wave = threadIdx.x >> 5;
    const int n = blockIdx.x * 8 + wave;

    float p[KFAC];
#pragma unroll
    for (int k = 0; k < KFAC; ++k) p[k] = 0.0f;

    const float* xrow = x + (size_t)n * DDIM;
    for (int d = lane; d < DDIM; d += 32) {
        float xv = xrow[d];
#pragma unroll
        for (int k = 0; k < KFAC; ++k) p[k] += xv * Wg[k * DDIM + d];
    }
#pragma unroll
    for (int k = 0; k < KFAC; ++k) {
        float v = p[k];
#pragma unroll
        for (int off = 16; off > 0; off >>= 1) v += __shfl_xor(v, off, 32);
        p[k] = v;
    }
    if (lane == 0) {
        float mx = -1e30f;
#pragma unroll
        for (int k = 0; k < KFAC; ++k) { p[k] += bg[k]; mx = fmaxf(mx, p[k]); }
        float s = 0.0f;
#pragma unroll
        for (int k = 0; k < KFAC; ++k) { p[k] = __expf(p[k] - mx); s += p[k]; }
        float inv = 1.0f / s, tot = 0.0f;
#pragma unroll
        for (int k = 0; k < KFAC; ++k) {
            float gv = p[k] * inv;
            g[n * KFAC + k] = gv;
            tot += gv;
        }
        atomicAdd(gsum, tot);
    }
}

// ---------------- facet balance scalar ----------------
__global__ void fb_kernel(const float* gsum, float* out_fb) {
    float m = (*gsum) / (float)(NDIM * KFAC);
    float d = m - 1.0f / (float)KFAC;
    *out_fb = d * d;
}

// ---------------- main fused kernel ----------------
// out[n,v] = sum_k sum_d (g[n,k]*x[n,d]) * Wf[k,v,d]  +  sum_k g[n,k]*bf[k,v]
// grid: (V/256, N/64); 256 threads = 8 waves as 4(M) x 2(V).
// Each wave: 16(M) x 128(V) strip = 8 WMMAs/chunk, single persistent accumulator,
// facets x d-chunks flattened into one 256-iteration double-buffered pipeline.
__global__ __launch_bounds__(256) void mfs_kernel(
    const float* __restrict__ x, const float* __restrict__ Wf,
    const float* __restrict__ bfb, const float* __restrict__ g,
    float* __restrict__ out)
{
    __shared__ alignas(16) float  ldsX[2 * XBUF];
    __shared__ alignas(16) bf16_t ldsW[2 * WBUFB];

    const int tid  = threadIdx.x;
    const int lane = tid & 31;
    const int wave = tid >> 5;
    const int wm   = wave >> 1;   // 0..3
    const int wv   = wave & 1;    // 0..1
    const int hf   = lane >> 4;   // 0/1
    const int l16  = lane & 15;

    const int n0 = blockIdx.y * MTILE;
    const int v0 = blockIdx.x * VTILE;

    // ---- x async staging descriptors (16B granules; facet 0 only) ----
    uint32_t xlds[2]; const float* xgp[2];
#pragma unroll
    for (int i = 0; i < 2; ++i) {
        int id = tid + 256 * i;          // 512 granules: 64 rows x 8
        int xr = id >> 3, xc = (id & 7) * 4;
        xlds[i] = (uint32_t)(uintptr_t)(ldsX + xr * XPAD + xc);
        xgp[i]  = x + (size_t)(n0 + xr) * DDIM + xc;
    }
    // ---- W register-staging descriptors (8-f32 granules) ----
    size_t wgo[4]; int wst[4];
#pragma unroll
    for (int i = 0; i < 4; ++i) {
        int id = tid + 256 * i;          // 1024 granules: 256 rows x 4
        int wr = id >> 2, wc = (id & 3) * 8;
        wgo[i] = (size_t)(v0 + wr) * DDIM + wc;
        wst[i] = wr * WPADB + wc;        // bf16 element offset in buffer
    }

    // ---- fragment LDS offsets ----
    const int aoff0 = (wm * 16 + l16) * XPAD + hf * 16;  // A: K = 16h+e
    const int aoff1 = aoff0 + 8;                         //    K = 16h+8+e
    int boff[8];
#pragma unroll
    for (int j = 0; j < 8; ++j)                          // B: N=l16, K=16h+e
        boff[j] = (wv * 128 + j * 16 + l16) * WPADB + hf * 16;

    const int growidx = (n0 + wm * 16 + l16) * KFAC;     // A-scale gate row

    v8f acc[8];
#pragma unroll
    for (int j = 0; j < 8; ++j) acc[j] = (v8f){};

    const size_t FSTRIDE = (size_t)VDIM * DDIM;

    // ---- prologue: stage chunk 0 ----
    v4f wreg[8];
#pragma unroll
    for (int i = 0; i < 4; ++i) {
        const float* p = Wf + wgo[i];
        wreg[2 * i]     = *(const v4f*)p;
        wreg[2 * i + 1] = *(const v4f*)(p + 4);
    }
#pragma unroll
    for (int i = 0; i < 2; ++i) async_ld16(xlds[i], xgp[i]);
#pragma unroll
    for (int i = 0; i < 4; ++i) {
        v8bf s;
#pragma unroll
        for (int e = 0; e < 4; ++e) {
            s[e]     = (bf16_t)wreg[2 * i][e];
            s[e + 4] = (bf16_t)wreg[2 * i + 1][e];
        }
        *(v8bf*)(ldsW + wst[i]) = s;
    }
    wait_async0();
    __syncthreads();

    for (int c = 0; c < TCHUNK; ++c) {
        const int cur = c & 1;
        const int nxt = cur ^ 1;
        const int cn  = c + 1;
        const bool more = cn < TCHUNK;

        // ---- issue global loads for chunk c+1 (hidden behind WMMAs) ----
        if (more) {
            const size_t base = (size_t)(cn >> 5) * FSTRIDE + (size_t)(cn & 31) * KCH;
#pragma unroll
            for (int i = 0; i < 4; ++i) {
                const float* p = Wf + base + wgo[i];
                wreg[2 * i]     = *(const v4f*)p;
                wreg[2 * i + 1] = *(const v4f*)(p + 4);
            }
            if (cn < NCHUNK) {   // x staged only during facet 0; persists after
#pragma unroll
                for (int i = 0; i < 2; ++i)
                    async_ld16(xlds[i] + (uint32_t)(nxt * XBUF * 4),
                               xgp[i] + (cn & 31) * KCH);
            }
        }

        // ---- compute chunk c: gate-scaled A, 8 WMMAs ----
        const float  gval = g[growidx + (c >> 5)];
        const float* xb   = ldsX + cur * XBUF;
        v8f a0 = *(const v8f*)(xb + aoff0);
        v8f a1 = *(const v8f*)(xb + aoff1);
        v16bf A = cvt16s(a0, a1, gval);
        const bf16_t* wb = ldsW + cur * WBUFB;
#pragma unroll
        for (int j = 0; j < 8; ++j) {
            v16bf B = *(const v16bf*)(wb + boff[j]);
            acc[j] = __builtin_amdgcn_wmma_f32_16x16x32_bf16(
                false, A, false, B, (short)0, acc[j], false, false);
        }

        // ---- convert + store chunk c+1 into the other buffer ----
        if (more) {
            bf16_t* wd = ldsW + nxt * WBUFB;
#pragma unroll
            for (int i = 0; i < 4; ++i) {
                v8bf s;
#pragma unroll
                for (int e = 0; e < 4; ++e) {
                    s[e]     = (bf16_t)wreg[2 * i][e];
                    s[e + 4] = (bf16_t)wreg[2 * i + 1][e];
                }
                *(v8bf*)(wd + wst[i]) = s;
            }
        }

        if (c < NCHUNK) wait_async0();   // x DMAs outstanding only in facet 0
        __syncthreads();
    }

    // ---- epilogue: add bias term sum_k g[n,k]*bf[k,v], then store ----
#pragma unroll
    for (int k = 0; k < KFAC; ++k) {
        float gr[8];
#pragma unroll
        for (int r = 0; r < 8; ++r)
            gr[r] = g[(n0 + wm * 16 + hf * 8 + r) * KFAC + k];
#pragma unroll
        for (int j = 0; j < 8; ++j) {
            float bfv = bfb[k * VDIM + v0 + wv * 128 + j * 16 + l16];
#pragma unroll
            for (int r = 0; r < 8; ++r)
                acc[j][r] += gr[r] * bfv;
        }
    }
#pragma unroll
    for (int j = 0; j < 8; ++j)
#pragma unroll
        for (int r = 0; r < 8; ++r)
            out[(size_t)(n0 + wm * 16 + hf * 8 + r) * VDIM
                + v0 + wv * 128 + j * 16 + l16] = acc[j][r];
}

extern "C" void kernel_launch(void* const* d_in, const int* in_sizes, int n_in,
                              void* d_out, int out_size, void* d_ws, size_t ws_size,
                              hipStream_t stream) {
    const float* x   = (const float*)d_in[0];
    const float* Wg  = (const float*)d_in[1];
    const float* bg  = (const float*)d_in[2];
    const float* Wf  = (const float*)d_in[3];
    const float* bfb = (const float*)d_in[4];
    float* out = (float*)d_out;

    float* g    = (float*)d_ws;        // N*KF floats
    float* gsum = g + NDIM * KFAC;     // 1 float

    init_ws_kernel<<<1, 32, 0, stream>>>(gsum);
    gate_kernel<<<NDIM / 8, 256, 0, stream>>>(x, Wg, bg, g, gsum);

    dim3 grid(VDIM / VTILE, NDIM / MTILE);
    mfs_kernel<<<grid, 256, 0, stream>>>(x, Wf, bfb, g, out);

    fb_kernel<<<1, 1, 0, stream>>>(gsum, out + (size_t)NDIM * VDIM);
}
